// TransformerBaseline_37374805410133
// MI455X (gfx1250) — compile-verified, matrix operand order checked by reading, not run on hardware
//
#include <hip/hip_runtime.h>
#include <cmath>

typedef _Float16 v16h __attribute__((ext_vector_type(16)));
typedef _Float16 v8h  __attribute__((ext_vector_type(8)));
typedef float    v8f  __attribute__((ext_vector_type(8)));

union H16 { v16h v; v8h h[2]; };

#define D_MODEL 256
#define N_LAYERS 6
#define N_HEADS 8
#define D_FF 1024
#define HEAD_DIM 32
#define SEQ_T 2048
#define BATCH 4
#define BT (BATCH * SEQ_T)   // 8192 tokens

#define DPP8_SEL(a,b,c,d,e,f,g,h) \
    ((a)|((b)<<3)|((c)<<6)|((d)<<9)|((e)<<12)|((f)<<15)|((g)<<18)|((h)<<21))

// max-reduce across the 16 lanes of each half-wave (lane ^1,^2,^4 via DPP8,
// lane ^8 via DPP16 row_ror:8 -- all VALU, no LDS hardware involved)
__device__ __forceinline__ float rowmax16_dpp(float v) {
    float t;
    t = __int_as_float(__builtin_amdgcn_mov_dpp8(__float_as_int(v),
            DPP8_SEL(1,0,3,2,5,4,7,6)));
    v = fmaxf(v, t);
    t = __int_as_float(__builtin_amdgcn_mov_dpp8(__float_as_int(v),
            DPP8_SEL(2,3,0,1,6,7,4,5)));
    v = fmaxf(v, t);
    t = __int_as_float(__builtin_amdgcn_mov_dpp8(__float_as_int(v),
            DPP8_SEL(4,5,6,7,0,1,2,3)));
    v = fmaxf(v, t);
    t = __int_as_float(__builtin_amdgcn_update_dpp(0, __float_as_int(v),
            0x128 /* row_ror:8 */, 0xF, 0xF, true));
    v = fmaxf(v, t);
    return v;
}

// async global->LDS copy of 16 bytes per lane (VGLOBAL op 98, ASYNCcnt path).
// INST_OFFSET is added to BOTH the LDS and global addresses (ISA 08 §4.4).
__device__ __forceinline__ void async_b128(unsigned ldsOff, const void* gaddr) {
    asm volatile("global_load_async_to_lds_b128 %0, %1, off"
                 :: "v"(ldsOff), "v"(gaddr) : "memory");
}
__device__ __forceinline__ void async_b128_o16(unsigned ldsOff, const void* gaddr) {
    asm volatile("global_load_async_to_lds_b128 %0, %1, off offset:16"
                 :: "v"(ldsOff), "v"(gaddr) : "memory");
}
__device__ __forceinline__ void wait_async0() {
    asm volatile("s_wait_asynccnt 0x0" ::: "memory");
}

// ---------------------------------------------------------------------------
// Copy f32 (init residual stream in d_out from input x)
// ---------------------------------------------------------------------------
__global__ void copy_f32(const float* __restrict__ src, float* __restrict__ dst) {
    size_t i = (size_t)blockIdx.x * blockDim.x + threadIdx.x;
    dst[i] = src[i];
}

// ---------------------------------------------------------------------------
// Transpose + convert: src f32 [K,N] row-major -> dst f16 [N,K] row-major
// ---------------------------------------------------------------------------
__global__ void conv_transpose_f16(const float* __restrict__ src, _Float16* __restrict__ dst,
                                   int K, int N) {
    __shared__ float tile[16][17];
    int n0 = blockIdx.x * 16, k0 = blockIdx.y * 16;
    tile[threadIdx.y][threadIdx.x] = src[(size_t)(k0 + threadIdx.y) * N + n0 + threadIdx.x];
    __syncthreads();
    dst[(size_t)(n0 + threadIdx.y) * K + k0 + threadIdx.x] = (_Float16)tile[threadIdx.x][threadIdx.y];
}

// ---------------------------------------------------------------------------
// LayerNorm over D=256, one block (256 threads) per token; f16 output
// ---------------------------------------------------------------------------
__global__ void ln_f16(const float* __restrict__ x, const float* __restrict__ s,
                       const float* __restrict__ b, _Float16* __restrict__ out) {
    int tok = blockIdx.x, i = threadIdx.x;
    float v = x[(size_t)tok * D_MODEL + i];
    __shared__ float r1[256], r2[256];
    r1[i] = v; r2[i] = v * v;
    __syncthreads();
    for (int o = 128; o > 0; o >>= 1) {
        if (i < o) { r1[i] += r1[i + o]; r2[i] += r2[i + o]; }
        __syncthreads();
    }
    float mu  = r1[0] * (1.0f / D_MODEL);
    float var = r2[0] * (1.0f / D_MODEL) - mu * mu;
    float nv  = (v - mu) * rsqrtf(var + 1e-5f) * s[i] + b[i];
    out[(size_t)tok * D_MODEL + i] = (_Float16)nv;
}

// ---------------------------------------------------------------------------
// WMMA GEMM: C[M,N] = epilogue(A[M,K] * Bt[N,K]^T + bias)
// Block = 256 threads (8 waves). Block tile 128(M) x 64(N).
// Double-buffered LDS fed by GLOBAL_LOAD_ASYNC_TO_LDS_B128 (ASYNCcnt):
// iteration i computes on buffer i&1 while the async engine fills i^1.
// Epilogues:
//  0: Q    -> f16 out = (acc+bias) * 1/sqrt(HEAD_DIM)
//  1: K    -> f16 out = acc+bias
//  2: VT   -> f16 out stored transposed per-head: Vt[b,h,d,t]
//  3: RES  -> f32 xres[row*N+col] += acc+bias      (out-proj / FF2 residual)
//  4: GELU -> f16 out = gelu_exact(acc+bias)       (FF1)
// ---------------------------------------------------------------------------
template <int EPI>
__global__ __launch_bounds__(256) void gemm_wmma(
    const _Float16* __restrict__ A,   // [M,K] f16
    const _Float16* __restrict__ Bt,  // [N,K] f16 (B transposed)
    const float*    __restrict__ bias,// [N]
    float*          __restrict__ xres,// residual f32 (EPI 3)
    _Float16*       __restrict__ out16,
    int M, int N, int K) {

    __shared__ _Float16 As[2][128 * 32];   // 2 x 8 KB
    __shared__ _Float16 Bs[2][64 * 32];    // 2 x 4 KB

    const int t    = threadIdx.x;
    const int w    = t >> 5;
    const int lane = t & 31;
    const int mrow = lane & 15;
    const int hi   = lane >> 4;           // 0 or 1
    const int klo  = hi ? 8 : 0;          // A/B fragment K sub-offset

    const int m0 = blockIdx.y * 128;
    const int n0 = blockIdx.x * 64;

    // cooperative tile fetch assignment (per thread)
    const int ar = t >> 1, ac = (t & 1) * 16;     // A: 32 bytes (2 x b128)
    const int bn = t >> 2, bc = (t & 3) * 8;      // B: 16 bytes (1 x b128)
    const _Float16* Ag = A  + (size_t)(m0 + ar) * K + ac;
    const _Float16* Bg = Bt + (size_t)(n0 + bn) * K + bc;
    const unsigned laA0 = (unsigned)(uintptr_t)&As[0][ar * 32 + ac];
    const unsigned laB0 = (unsigned)(uintptr_t)&Bs[0][bn * 32 + bc];
    const unsigned laA1 = (unsigned)(uintptr_t)&As[1][ar * 32 + ac];
    const unsigned laB1 = (unsigned)(uintptr_t)&Bs[1][bn * 32 + bc];

    v8f acc[4];
    #pragma unroll
    for (int s = 0; s < 4; ++s) acc[s] = (v8f){};

    // prologue: async-fill buffer 0 with the k0=0 tile
    async_b128(laA0, Ag);
    async_b128_o16(laA0, Ag);
    async_b128(laB0, Bg);

    const int nk = K >> 5;
    for (int i = 0; i < nk; ++i) {
        const int buf = i & 1;
        wait_async0();          // only this tile's 3 ops are outstanding
        __syncthreads();        // tile visible block-wide; prev buffer reads done

        if (i + 1 < nk) {       // prefetch next tile into the other buffer
            const _Float16* Agn = Ag + (i + 1) * 32;
            const _Float16* Bgn = Bg + (i + 1) * 32;
            async_b128(buf ? laA0 : laA1, Agn);
            async_b128_o16(buf ? laA0 : laA1, Agn);
            async_b128(buf ? laB0 : laB1, Bgn);
        }

        H16 af;
        af.h[0] = *(const v8h*)(&As[buf][(w * 16 + mrow) * 32 + klo]);
        af.h[1] = *(const v8h*)(&As[buf][(w * 16 + mrow) * 32 + klo + 16]);
        #pragma unroll
        for (int s = 0; s < 4; ++s) {
            H16 bf;
            bf.h[0] = *(const v8h*)(&Bs[buf][(s * 16 + mrow) * 32 + klo]);
            bf.h[1] = *(const v8h*)(&Bs[buf][(s * 16 + mrow) * 32 + klo + 16]);
            acc[s] = __builtin_amdgcn_wmma_f32_16x16x32_f16(
                false, af.v, false, bf.v, (short)0, acc[s], false, false);
        }
    }

    // epilogue
    #pragma unroll
    for (int s = 0; s < 4; ++s) {
        #pragma unroll
        for (int r = 0; r < 8; ++r) {
            int row = m0 + w * 16 + r + hi * 8;
            int col = n0 + s * 16 + mrow;
            float v = acc[s][r] + bias[col];
            if (EPI == 0) {
                out16[(size_t)row * N + col] = (_Float16)(v * 0.17677669529663687f);
            } else if (EPI == 1) {
                out16[(size_t)row * N + col] = (_Float16)v;
            } else if (EPI == 2) {
                int b = row >> 11, tt = row & (SEQ_T - 1);
                int h = col >> 5, d = col & 31;
                out16[(((size_t)(b * N_HEADS + h) * HEAD_DIM + d) << 11) + tt] = (_Float16)v;
            } else if (EPI == 3) {
                xres[(size_t)row * N + col] += v;
            } else { // GELU exact
                float g = 0.5f * v * (1.0f + erff(v * 0.70710678118654752f));
                out16[(size_t)row * N + col] = (_Float16)g;
            }
        }
    }
}

// ---------------------------------------------------------------------------
// Flash attention, non-causal. Grid (T/128, B*H), 256 threads = 8 waves.
// Wave handles 16 query rows; online softmax over 64-token K chunks.
// Scale pre-folded into Q. Row max via DPP (pure VALU); row sum via a
// P @ ones WMMA (result lands broadcast across each 16-lane half in C layout).
// ---------------------------------------------------------------------------
__global__ __launch_bounds__(256) void attn_wmma(
    const _Float16* __restrict__ q16,  // [B,T,D] f16 (scaled)
    const _Float16* __restrict__ k16,  // [B,T,D] f16
    const _Float16* __restrict__ vt16, // [B,H,HD,T] f16
    _Float16*       __restrict__ o16)  // [B,T,D] f16
{
    __shared__ _Float16 Plds[8 * 16 * 64];   // per-wave 16x64 P tile (2 KB each)

    const int t    = threadIdx.x;
    const int w    = t >> 5;
    const int lane = t & 31;
    const int mrow = lane & 15;
    const int hi   = lane >> 4;
    const int klo  = hi ? 8 : 0;

    const int bh = blockIdx.y;
    const int b  = bh >> 3;
    const int h  = bh & 7;
    const int qbase = blockIdx.x * 128 + w * 16;
    const size_t tokBase = (size_t)b * SEQ_T;

    // Q fragment (A operand, 16x32), scale already folded in
    H16 qa;
    {
        const _Float16* qp = q16 + ((tokBase + qbase + mrow) * D_MODEL + h * HEAD_DIM);
        qa.h[0] = *(const v8h*)(qp + klo);
        qa.h[1] = *(const v8h*)(qp + klo + 16);
    }

    // all-ones B operand for row-sum matmul (layout independent)
    H16 ones;
    #pragma unroll
    for (int i = 0; i < 8; ++i) { ones.h[0][i] = (_Float16)1.0f; ones.h[1][i] = (_Float16)1.0f; }

    float mreg[8], lreg[8], corr[8];
    v8f o0 = (v8f){}, o1 = (v8f){};
    #pragma unroll
    for (int r = 0; r < 8; ++r) { mreg[r] = -3.0e38f; lreg[r] = 0.0f; }

    _Float16* Pw = &Plds[w * 16 * 64];
    const _Float16* vbase = vt16 + (size_t)bh * HEAD_DIM * SEQ_T;

    for (int j = 0; j < SEQ_T; j += 64) {
        // ---- S = Q K^T for 64 k-tokens: four 16x16 tiles -------------------
        v8f sS[4];
        v8f zc = (v8f){};
        #pragma unroll
        for (int kk = 0; kk < 4; ++kk) {
            H16 kb;
            const _Float16* kp = k16 + ((tokBase + j + kk * 16 + mrow) * D_MODEL + h * HEAD_DIM);
            kb.h[0] = *(const v8h*)(kp + klo);
            kb.h[1] = *(const v8h*)(kp + klo + 16);
            sS[kk] = __builtin_amdgcn_wmma_f32_16x16x32_f16(
                false, qa.v, false, kb.v, (short)0, zc, false, false);
        }

        // ---- online softmax: one cross-lane max pass per 64 tokens ---------
        #pragma unroll
        for (int r = 0; r < 8; ++r) {
            float mx = fmaxf(fmaxf(sS[0][r], sS[1][r]), fmaxf(sS[2][r], sS[3][r]));
            mx = rowmax16_dpp(mx);
            float mnew = fmaxf(mreg[r], mx);
            corr[r] = __expf(mreg[r] - mnew);
            mreg[r] = mnew;
            o0[r] *= corr[r];
            o1[r] *= corr[r];
            int prow = r + hi * 8;
            #pragma unroll
            for (int kk = 0; kk < 4; ++kk) {
                float p = __expf(sS[kk][r] - mnew);
                Pw[prow * 64 + kk * 16 + mrow] = (_Float16)p;
            }
        }

        // ---- P fragments (A operand halves: tokens 0-31, 32-63) ------------
        H16 pa0, pa1;
        pa0.h[0] = *(const v8h*)(&Pw[mrow * 64 + klo]);
        pa0.h[1] = *(const v8h*)(&Pw[mrow * 64 + klo + 16]);
        pa1.h[0] = *(const v8h*)(&Pw[mrow * 64 + 32 + klo]);
        pa1.h[1] = *(const v8h*)(&Pw[mrow * 64 + 32 + klo + 16]);

        // ---- row sums via P @ ones (replaces all sum shuffles) -------------
        v8f sums = __builtin_amdgcn_wmma_f32_16x16x32_f16(
            false, pa0.v, false, ones.v, (short)0, zc, false, false);
        sums = __builtin_amdgcn_wmma_f32_16x16x32_f16(
            false, pa1.v, false, ones.v, (short)0, sums, false, false);
        #pragma unroll
        for (int r = 0; r < 8; ++r) lreg[r] = lreg[r] * corr[r] + sums[r];

        // ---- O += P V (V transposed: contiguous along tokens) --------------
        #pragma unroll
        for (int kk = 0; kk < 2; ++kk) {
            const int joff = j + kk * 32;
            H16 vb0, vb1;
            const _Float16* vp0 = vbase + (size_t)mrow * SEQ_T + joff;
            vb0.h[0] = *(const v8h*)(vp0 + klo);
            vb0.h[1] = *(const v8h*)(vp0 + klo + 16);
            const _Float16* vp1 = vbase + (size_t)(16 + mrow) * SEQ_T + joff;
            vb1.h[0] = *(const v8h*)(vp1 + klo);
            vb1.h[1] = *(const v8h*)(vp1 + klo + 16);
            const v16h& pav = kk ? pa1.v : pa0.v;
            o0 = __builtin_amdgcn_wmma_f32_16x16x32_f16(
                false, pav, false, vb0.v, (short)0, o0, false, false);
            o1 = __builtin_amdgcn_wmma_f32_16x16x32_f16(
                false, pav, false, vb1.v, (short)0, o1, false, false);
        }
    }

    // finalize: divide by l, write f16 [B,T,D]
    #pragma unroll
    for (int r = 0; r < 8; ++r) {
        int row = qbase + r + hi * 8;
        float inv = 1.0f / lreg[r];
        size_t base = (tokBase + row) * D_MODEL + h * HEAD_DIM;
        o16[base + mrow]      = (_Float16)(o0[r] * inv);
        o16[base + 16 + mrow] = (_Float16)(o1[r] * inv);
    }
}

// ---------------------------------------------------------------------------
// Host-side orchestration
// ---------------------------------------------------------------------------
extern "C" void kernel_launch(void* const* d_in, const int* in_sizes, int n_in,
                              void* d_out, int out_size, void* d_ws, size_t ws_size,
                              hipStream_t stream) {
    const float* x_in  = (const float*)d_in[0];
    const float* ln1_s = (const float*)d_in[1];
    const float* ln1_b = (const float*)d_in[2];
    const float* wq = (const float*)d_in[3];
    const float* bq = (const float*)d_in[4];
    const float* wk = (const float*)d_in[5];
    const float* bk = (const float*)d_in[6];
    const float* wv = (const float*)d_in[7];
    const float* bv = (const float*)d_in[8];
    const float* wo = (const float*)d_in[9];
    const float* bo = (const float*)d_in[10];
    const float* ln2_s = (const float*)d_in[11];
    const float* ln2_b = (const float*)d_in[12];
    const float* w1 = (const float*)d_in[13];
    const float* b1 = (const float*)d_in[14];
    const float* w2 = (const float*)d_in[15];
    const float* b2 = (const float*)d_in[16];

    float* x = (float*)d_out;   // running residual stream [BT, D]

    // workspace bump allocator (256B aligned)
    char* wsp = (char*)d_ws;
    auto alloc = [&](size_t bytes) -> void* {
        void* p = (void*)wsp;
        wsp += (bytes + 255) & ~(size_t)255;
        return p;
    };
    const size_t WDD = (size_t)D_MODEL * D_MODEL;          // 65536
    const size_t WDF = (size_t)D_MODEL * D_FF;             // 262144
    _Float16* wqT = (_Float16*)alloc(N_LAYERS * WDD * 2);
    _Float16* wkT = (_Float16*)alloc(N_LAYERS * WDD * 2);
    _Float16* wvT = (_Float16*)alloc(N_LAYERS * WDD * 2);
    _Float16* woT = (_Float16*)alloc(N_LAYERS * WDD * 2);
    _Float16* w1T = (_Float16*)alloc(N_LAYERS * WDF * 2);  // [D_FF, D_MODEL]
    _Float16* w2T = (_Float16*)alloc(N_LAYERS * WDF * 2);  // [D_MODEL, D_FF]
    _Float16* n16 = (_Float16*)alloc((size_t)BT * D_MODEL * 2);
    _Float16* q16 = (_Float16*)alloc((size_t)BT * D_MODEL * 2);
    _Float16* k16 = (_Float16*)alloc((size_t)BT * D_MODEL * 2);
    _Float16* vt16 = (_Float16*)alloc((size_t)BT * D_MODEL * 2);  // [B,H,HD,T]
    _Float16* a16 = (_Float16*)alloc((size_t)BT * D_MODEL * 2);
    _Float16* h16 = (_Float16*)alloc((size_t)BT * D_FF * 2);

    // init residual stream
    copy_f32<<<(BT * D_MODEL) / 256, 256, 0, stream>>>(x_in, x);

    // weight conversion + transpose to f16
    dim3 tB(16, 16);
    for (int l = 0; l < N_LAYERS; ++l) {
        conv_transpose_f16<<<dim3(D_MODEL/16, D_MODEL/16), tB, 0, stream>>>(wq + l*WDD, wqT + l*WDD, D_MODEL, D_MODEL);
        conv_transpose_f16<<<dim3(D_MODEL/16, D_MODEL/16), tB, 0, stream>>>(wk + l*WDD, wkT + l*WDD, D_MODEL, D_MODEL);
        conv_transpose_f16<<<dim3(D_MODEL/16, D_MODEL/16), tB, 0, stream>>>(wv + l*WDD, wvT + l*WDD, D_MODEL, D_MODEL);
        conv_transpose_f16<<<dim3(D_MODEL/16, D_MODEL/16), tB, 0, stream>>>(wo + l*WDD, woT + l*WDD, D_MODEL, D_MODEL);
        conv_transpose_f16<<<dim3(D_FF/16,    D_MODEL/16), tB, 0, stream>>>(w1 + l*WDF, w1T + l*WDF, D_MODEL, D_FF);
        conv_transpose_f16<<<dim3(D_MODEL/16, D_FF/16),    tB, 0, stream>>>(w2 + l*WDF, w2T + l*WDF, D_FF, D_MODEL);
    }

    dim3 gD(D_MODEL / 64, BT / 128);   // N=256 GEMMs
    dim3 gF(D_FF / 64,    BT / 128);   // N=1024 GEMM (FF1)

    for (int l = 0; l < N_LAYERS; ++l) {
        // LN1 -> f16
        ln_f16<<<BT, 256, 0, stream>>>(x, ln1_s + l*D_MODEL, ln1_b + l*D_MODEL, n16);
        // Q (scale folded), K, V (stored transposed per head)
        gemm_wmma<0><<<gD, 256, 0, stream>>>(n16, wqT + l*WDD, bq + l*D_MODEL, nullptr, q16,  BT, D_MODEL, D_MODEL);
        gemm_wmma<1><<<gD, 256, 0, stream>>>(n16, wkT + l*WDD, bk + l*D_MODEL, nullptr, k16,  BT, D_MODEL, D_MODEL);
        gemm_wmma<2><<<gD, 256, 0, stream>>>(n16, wvT + l*WDD, bv + l*D_MODEL, nullptr, vt16, BT, D_MODEL, D_MODEL);
        // flash attention
        attn_wmma<<<dim3(SEQ_T / 128, BATCH * N_HEADS), 256, 0, stream>>>(q16, k16, vt16, a16);
        // out projection + residual into x
        gemm_wmma<3><<<gD, 256, 0, stream>>>(a16, woT + l*WDD, bo + l*D_MODEL, x, nullptr, BT, D_MODEL, D_MODEL);
        // LN2 -> f16
        ln_f16<<<BT, 256, 0, stream>>>(x, ln2_s + l*D_MODEL, ln2_b + l*D_MODEL, n16);
        // FF1 + exact GELU -> f16
        gemm_wmma<4><<<gF, 256, 0, stream>>>(n16, w1T + l*WDF, b1 + l*D_FF, nullptr, h16, BT, D_FF, D_MODEL);
        // FF2 + residual into x
        gemm_wmma<3><<<gD, 256, 0, stream>>>(h16, w2T + l*WDF, b2 + l*D_MODEL, x, nullptr, BT, D_MODEL, D_FF);
    }
}
